// Attention_35871566856924
// MI455X (gfx1250) — matrix operand order repaired and checked by reading, not compiled, not run on hardware
//
#include <hip/hip_runtime.h>

// ---------------------------------------------------------------------------
// CvT attention block for MI455X (gfx1250, wave32, WMMA).
// All matmuls use v_wmma_f32_16x16x32_bf16. Flash-attention inner loop.
// ---------------------------------------------------------------------------

typedef __attribute__((ext_vector_type(16))) __bf16 v16bf;
typedef __attribute__((ext_vector_type(8)))  float  v8f;

union BFVec { v16bf v; unsigned int u[8]; };

__device__ __forceinline__ unsigned short bf16_rne(float f) {
  unsigned u = __float_as_uint(f);
  u += 0x7FFFu + ((u >> 16) & 1u);           // round-to-nearest-even
  return (unsigned short)(u >> 16);
}

__device__ __forceinline__ v8f wmma_bf16(v16bf a, v16bf b, v8f c) {
  return __builtin_amdgcn_wmma_f32_16x16x32_bf16(false, a, false, b, (short)0, c,
                                                 false, false);
}

// A-tile 16(M)x32(K) bf16 from row-major src (ld elements), origin (row0,k0).
// lanes 0-15 row=lane hold K 0..15; lanes 16-31 row=lane-16
// hold K 16..31; each VGPR = 2 consecutive K -> one b32 load each.
__device__ __forceinline__ v16bf load_a_tile(const unsigned short* src, int ld,
                                             int row0, int k0, int lane) {
  int hx = lane >> 4, m = lane & 15;
  BFVec a;
  const unsigned short* p = src + (size_t)(row0 + m) * ld + k0;
#pragma unroll
  for (int i = 0; i < 8; ++i) {
    int k = hx * 16 + i * 2;
    a.u[i] = *(const unsigned int*)(p + k);
  }
  return a.v;
}

// B-tile 32(K)x16(N) bf16 where element (k,n) = src[(n0+n)*ld + k0 + k]
// (source N-major, K contiguous). Lanes 0-15 col=lane hold K 0..15,
// lanes 16-31 hold K 16..31 -> 8 contiguous b32 loads per lane.
__device__ __forceinline__ v16bf load_b_tile(const unsigned short* src, int ld,
                                             int n0, int k0, int lane) {
  int hx = lane >> 4, n = lane & 15;
  BFVec b;
  const unsigned short* p = src + (size_t)(n0 + n) * ld + k0 + hx * 16;
#pragma unroll
  for (int i = 0; i < 8; ++i) b.u[i] = *(const unsigned int*)(p + 2 * i);
  return b.v;
}

// ---------------------------------------------------------------------------
// fp32 -> bf16 weight conversion
__global__ void f2bf_kernel(const float* __restrict__ in,
                            unsigned short* __restrict__ out, int n) {
  int i = blockIdx.x * blockDim.x + threadIdx.x;
  if (i < n) out[i] = bf16_rne(in[i]);
}

// ---------------------------------------------------------------------------
// depthwise 3x3 conv + BatchNorm(eval) fused, output bf16 stored [b][n][c]
// (channel-contiguous so the following 1x1-conv GEMM B-tiles are contiguous).
__global__ void dwbn_kernel(const float* __restrict__ x, const float* __restrict__ wdw,
                            const float* __restrict__ g, const float* __restrict__ bb,
                            const float* __restrict__ mu, const float* __restrict__ var,
                            unsigned short* __restrict__ y, int Hout, int Wout,
                            int stride) {
  int idx = blockIdx.x * blockDim.x + threadIdx.x;
  int total = 8 * 256 * Hout * Wout;
  if (idx >= total) return;
  int wo = idx % Wout; int t = idx / Wout;
  int ho = t % Hout;   t /= Hout;
  int c  = t % 256;    int b = t / 256;
  const float* xp = x + (size_t)(b * 256 + c) * 56 * 56;
  const float* wp = wdw + c * 9;
  float acc = 0.f;
  int hi0 = ho * stride - 1, wi0 = wo * stride - 1;
#pragma unroll
  for (int kh = 0; kh < 3; ++kh) {
    int hi = hi0 + kh;
    if ((unsigned)hi >= 56u) continue;
#pragma unroll
    for (int kw = 0; kw < 3; ++kw) {
      int wi = wi0 + kw;
      if ((unsigned)wi >= 56u) continue;
      acc += xp[hi * 56 + wi] * wp[kh * 3 + kw];
    }
  }
  float inv = g[c] * rsqrtf(var[c] + 1e-5f);
  float val = acc * inv + (bb[c] - mu[c] * inv);
  int n = ho * Wout + wo;
  y[((size_t)b * (Hout * Wout) + n) * 256 + c] = bf16_rne(val);
}

// ---------------------------------------------------------------------------
// Q projection GEMM: Q[co,n] = Wq[co,ci] * Yq[n,ci]^T   (per batch)
// One wave -> 64(M) x 16(N) tile, K=256 in 8 steps, 4 WMMAs/step (B reuse).
// Store Q as [b*4+h][n][d] bf16 (d contiguous for attention A/B tiles).
__global__ void gemm_q_kernel(const unsigned short* __restrict__ W,   // [256][256]
                              const unsigned short* __restrict__ Y,   // [8][3136][256]
                              unsigned short* __restrict__ Q) {       // [32][3136][64]
  int lane = threadIdx.x & 31;
  int n0 = blockIdx.x * 16;   // 196 n-tiles
  int m0 = blockIdx.y * 64;   // 4 m-blocks
  int b  = blockIdx.z;
  const unsigned short* Yp = Y + (size_t)b * 3136 * 256;
  v8f acc[4] = {};
  for (int k0 = 0; k0 < 256; k0 += 32) {
    v16bf bt = load_b_tile(Yp, 256, n0, k0, lane);
#pragma unroll
    for (int t = 0; t < 4; ++t) {
      v16bf at = load_a_tile(W, 256, m0 + t * 16, k0, lane);
      acc[t] = wmma_bf16(at, bt, acc[t]);
    }
  }
  int hx = lane >> 4, nc = lane & 15;
  int n = n0 + nc;
#pragma unroll
  for (int t = 0; t < 4; ++t)
#pragma unroll
    for (int i = 0; i < 8; ++i) {
      int co = m0 + t * 16 + hx * 8 + i;
      int h = co >> 6, d = co & 63;
      Q[(((size_t)b * 4 + h) * 3136 + n) * 64 + d] = bf16_rne(acc[t][i]);
    }
}

// KV projection GEMM: [512,ci] weights; K -> [bh][n(pad800)][d], V -> [bh][d][n(pad800)]
__global__ void gemm_kv_kernel(const unsigned short* __restrict__ W,   // [512][256]
                               const unsigned short* __restrict__ Y,   // [8][784][256]
                               unsigned short* __restrict__ Kb,        // [32][800][64]
                               unsigned short* __restrict__ Vb) {      // [32][64][800]
  int lane = threadIdx.x & 31;
  int n0 = blockIdx.x * 16;   // 49 n-tiles
  int m0 = blockIdx.y * 64;   // 8 m-blocks
  int b  = blockIdx.z;
  const unsigned short* Yp = Y + (size_t)b * 784 * 256;
  v8f acc[4] = {};
  for (int k0 = 0; k0 < 256; k0 += 32) {
    v16bf bt = load_b_tile(Yp, 256, n0, k0, lane);
#pragma unroll
    for (int t = 0; t < 4; ++t) {
      v16bf at = load_a_tile(W, 256, m0 + t * 16, k0, lane);
      acc[t] = wmma_bf16(at, bt, acc[t]);
    }
  }
  int hx = lane >> 4, nc = lane & 15;
  int n = n0 + nc;
#pragma unroll
  for (int t = 0; t < 4; ++t)
#pragma unroll
    for (int i = 0; i < 8; ++i) {
      int co = m0 + t * 16 + hx * 8 + i;
      if (co < 256) {
        int h = co >> 6, d = co & 63;
        Kb[(((size_t)b * 4 + h) * 800 + n) * 64 + d] = bf16_rne(acc[t][i]);
      } else {
        int c2 = co - 256; int h = c2 >> 6, d = c2 & 63;
        Vb[(((size_t)b * 4 + h) * 64 + d) * 800 + n] = bf16_rne(acc[t][i]);
      }
    }
}

// zero the KV pad region n in [784,800)
__global__ void pad_kernel(unsigned short* __restrict__ Kb,
                           unsigned short* __restrict__ Vb) {
  int idx = blockIdx.x * blockDim.x + threadIdx.x;   // 32*16*64
  if (idx >= 32768) return;
  int d = idx & 63; int t = idx >> 6;
  int n = 784 + (t & 15); int bh = t >> 4;
  Kb[((size_t)bh * 800 + n) * 64 + d] = 0;
  Vb[((size_t)bh * 64 + d) * 800 + n] = 0;
}

// ---------------------------------------------------------------------------
// Flash attention: per wave a 16-query tile; loop KV in chunks of 32 with
// online softmax. 8 WMMAs per iteration (4 QK^T + 4 AV). P routed through LDS
// (C-layout -> A-layout transpose). Output stored [b][n][c] bf16 for out-proj.
__global__ __launch_bounds__(128) void attn_kernel(
    const unsigned short* __restrict__ Qb,   // [32][3136][64]
    const unsigned short* __restrict__ Kb,   // [32][800][64]
    const unsigned short* __restrict__ Vb,   // [32][64][800]
    unsigned short* __restrict__ Ob) {       // [8][3136][256]
  __shared__ __align__(16) unsigned short pbuf[4][16][32];
  int wave = threadIdx.x >> 5;
  int lane = threadIdx.x & 31;
  int hx = lane >> 4, lc = lane & 15;
  int bh = blockIdx.y;                 // 0..31
  int b = bh >> 2, h = bh & 3;
  int n0 = (blockIdx.x * 4 + wave) * 16;   // query tile origin (196 tiles)
  const unsigned short* Qp = Qb + ((size_t)bh * 3136 + n0) * 64;
  const unsigned short* Kp = Kb + (size_t)bh * 800 * 64;
  const unsigned short* Vp = Vb + (size_t)bh * 64 * 800;

  v16bf qa0 = load_a_tile(Qp, 64, 0, 0, lane);     // d = 0..31
  v16bf qa1 = load_a_tile(Qp, 64, 0, 32, lane);    // d = 32..63

  float mrow[8], lrow[8];
  v8f o[4] = {};
#pragma unroll
  for (int i = 0; i < 8; ++i) { mrow[i] = -1e30f; lrow[i] = 0.f; }

  for (int j0 = 0; j0 < 800; j0 += 32) {
    float p0[8], p1[8];
    {   // S tile, columns j0..j0+15
      v16bf kb0 = load_b_tile(Kp, 64, j0, 0, lane);
      v16bf kb1 = load_b_tile(Kp, 64, j0, 32, lane);
      v8f s = {};
      s = wmma_bf16(qa0, kb0, s);
      s = wmma_bf16(qa1, kb1, s);
      int jc = j0 + lc;
#pragma unroll
      for (int i = 0; i < 8; ++i) p0[i] = (jc < 784) ? s[i] * 0.125f : -1e30f;
    }
    {   // S tile, columns j0+16..j0+31
      v16bf kb0 = load_b_tile(Kp, 64, j0 + 16, 0, lane);
      v16bf kb1 = load_b_tile(Kp, 64, j0 + 16, 32, lane);
      v8f s = {};
      s = wmma_bf16(qa0, kb0, s);
      s = wmma_bf16(qa1, kb1, s);
      int jc = j0 + 16 + lc;
#pragma unroll
      for (int i = 0; i < 8; ++i) p1[i] = (jc < 784) ? s[i] * 0.125f : -1e30f;
    }
    // online softmax: row reductions over the 16-lane half (rows hx*8+i)
    float alpha[8];
#pragma unroll
    for (int i = 0; i < 8; ++i) {
      float v = fmaxf(p0[i], p1[i]);
      v = fmaxf(v, __shfl_xor(v, 1, 32));
      v = fmaxf(v, __shfl_xor(v, 2, 32));
      v = fmaxf(v, __shfl_xor(v, 4, 32));
      v = fmaxf(v, __shfl_xor(v, 8, 32));
      float mn = fmaxf(mrow[i], v);
      alpha[i] = __expf(mrow[i] - mn);
      p0[i] = __expf(p0[i] - mn);
      p1[i] = __expf(p1[i] - mn);
      float rs = p0[i] + p1[i];
      rs += __shfl_xor(rs, 1, 32);
      rs += __shfl_xor(rs, 2, 32);
      rs += __shfl_xor(rs, 4, 32);
      rs += __shfl_xor(rs, 8, 32);
      lrow[i] = lrow[i] * alpha[i] + rs;
      mrow[i] = mn;
    }
#pragma unroll
    for (int t = 0; t < 4; ++t)
#pragma unroll
      for (int i = 0; i < 8; ++i) o[t][i] *= alpha[i];
    // C-layout P -> LDS (row-major 16x32)
#pragma unroll
    for (int i = 0; i < 8; ++i) {
      int r = hx * 8 + i;
      pbuf[wave][r][lc]      = bf16_rne(p0[i]);
      pbuf[wave][r][16 + lc] = bf16_rne(p1[i]);
    }
    __syncthreads();
    v16bf pa = load_a_tile(&pbuf[wave][0][0], 32, 0, 0, lane);
#pragma unroll
    for (int t = 0; t < 4; ++t) {
      v16bf vt = load_b_tile(Vp, 800, t * 16, j0, lane);   // N=d, K=j contiguous
      o[t] = wmma_bf16(pa, vt, o[t]);
    }
    __syncthreads();
  }
  // epilogue: normalize and store [b][n][c]
#pragma unroll
  for (int t = 0; t < 4; ++t)
#pragma unroll
    for (int i = 0; i < 8; ++i) {
      int n = n0 + hx * 8 + i;
      int c = h * 64 + t * 16 + lc;
      Ob[((size_t)b * 3136 + n) * 256 + c] = bf16_rne(o[t][i] / lrow[i]);
    }
}

// ---------------------------------------------------------------------------
// Output projection: out[co,n] = Wo[co,ci] * O[n,ci]^T + bo[co], fp32 out.
__global__ void gemm_out_kernel(const unsigned short* __restrict__ W,  // [256][256]
                                const unsigned short* __restrict__ Y,  // [8][3136][256]
                                const float* __restrict__ bias,
                                float* __restrict__ out) {             // [8][256][3136]
  int lane = threadIdx.x & 31;
  int n0 = blockIdx.x * 16;
  int m0 = blockIdx.y * 64;
  int b  = blockIdx.z;
  const unsigned short* Yp = Y + (size_t)b * 3136 * 256;
  v8f acc[4] = {};
  for (int k0 = 0; k0 < 256; k0 += 32) {
    v16bf bt = load_b_tile(Yp, 256, n0, k0, lane);
#pragma unroll
    for (int t = 0; t < 4; ++t) {
      v16bf at = load_a_tile(W, 256, m0 + t * 16, k0, lane);
      acc[t] = wmma_bf16(at, bt, acc[t]);
    }
  }
  int hx = lane >> 4, nc = lane & 15;
  int n = n0 + nc;
#pragma unroll
  for (int t = 0; t < 4; ++t)
#pragma unroll
    for (int i = 0; i < 8; ++i) {
      int co = m0 + t * 16 + hx * 8 + i;
      out[((size_t)b * 256 + co) * 3136 + n] = acc[t][i] + bias[co];
    }
}

// ---------------------------------------------------------------------------
extern "C" void kernel_launch(void* const* d_in, const int* in_sizes, int n_in,
                              void* d_out, int out_size, void* d_ws, size_t ws_size,
                              hipStream_t stream) {
  (void)in_sizes; (void)n_in; (void)out_size; (void)ws_size;
  const float* x      = (const float*)d_in[0];
  const float* wq_dw  = (const float*)d_in[1];
  const float* wq_pw  = (const float*)d_in[2];
  const float* wkv_dw = (const float*)d_in[3];
  const float* wkv_pw = (const float*)d_in[4];
  const float* wo     = (const float*)d_in[5];
  const float* bo     = (const float*)d_in[6];
  const float* bnq_g  = (const float*)d_in[7];
  const float* bnq_b  = (const float*)d_in[8];
  const float* bnq_m  = (const float*)d_in[9];
  const float* bnq_v  = (const float*)d_in[10];
  const float* bnkv_g = (const float*)d_in[11];
  const float* bnkv_b = (const float*)d_in[12];
  const float* bnkv_m = (const float*)d_in[13];
  const float* bnkv_v = (const float*)d_in[14];
  float* out = (float*)d_out;

  char* ws = (char*)d_ws;
  size_t off = 0;
  unsigned short* wq_bf  = (unsigned short*)(ws + off); off += (size_t)256*256*2;
  unsigned short* wkv_bf = (unsigned short*)(ws + off); off += (size_t)512*256*2;
  unsigned short* wo_bf  = (unsigned short*)(ws + off); off += (size_t)256*256*2;
  unsigned short* yq     = (unsigned short*)(ws + off); off += (size_t)8*3136*256*2;
  unsigned short* ykv    = (unsigned short*)(ws + off); off += (size_t)8*784*256*2;
  unsigned short* qb     = (unsigned short*)(ws + off); off += (size_t)32*3136*64*2;
  unsigned short* kb     = (unsigned short*)(ws + off); off += (size_t)32*800*64*2;
  unsigned short* vb     = (unsigned short*)(ws + off); off += (size_t)32*800*64*2;
  unsigned short* ob     = (unsigned short*)(ws + off); off += (size_t)8*3136*256*2;

  f2bf_kernel<<<(65536 + 255) / 256, 256, 0, stream>>>(wq_pw, wq_bf, 65536);
  f2bf_kernel<<<(131072 + 255) / 256, 256, 0, stream>>>(wkv_pw, wkv_bf, 131072);
  f2bf_kernel<<<(65536 + 255) / 256, 256, 0, stream>>>(wo, wo_bf, 65536);

  int tq = 8 * 256 * 56 * 56;
  dwbn_kernel<<<(tq + 255) / 256, 256, 0, stream>>>(x, wq_dw, bnq_g, bnq_b, bnq_m,
                                                    bnq_v, yq, 56, 56, 1);
  int tkv = 8 * 256 * 28 * 28;
  dwbn_kernel<<<(tkv + 255) / 256, 256, 0, stream>>>(x, wkv_dw, bnkv_g, bnkv_b,
                                                     bnkv_m, bnkv_v, ykv, 28, 28, 2);

  gemm_q_kernel<<<dim3(196, 4, 8), 32, 0, stream>>>(wq_bf, yq, qb);
  gemm_kv_kernel<<<dim3(49, 8, 8), 32, 0, stream>>>(wkv_bf, ykv, kb, vb);
  pad_kernel<<<(32768 + 255) / 256, 256, 0, stream>>>(kb, vb);
  attn_kernel<<<dim3(49, 32), 128, 0, stream>>>(qb, kb, vb, ob);
  gemm_out_kernel<<<dim3(196, 4, 8), 32, 0, stream>>>(wo_bf, ob, bo, out);
}
